// FoundationStereoOnnx_47150150975649
// MI455X (gfx1250) — matrix operand order repaired and checked by reading, not compiled, not run on hardware
//
#include <hip/hip_runtime.h>
#include <hip/hip_bf16.h>

// ---------------------------------------------------------------------------
// FoundationStereo-style refinement net on gfx1250 (CDNA5).
//  - every conv / correlation GEMM runs on v_wmma_f32_16x16x32_f16 (f32 acc)
//  - each conv wave computes a 32-pixel x 64-channel tile (8 accumulators):
//    A fragments reused across 4 N-subtiles, B fragments across 2 M-subtiles
//  - weight / correlation B tiles are DMA'd into LDS by the Tensor Data Mover
//    (tensor_load_to_lds), double-buffered with s_wait_tensorcnt
//  - im2col K-decomposition is incremental (2 divisions per 8-K run, 0 for 1x1)
// ---------------------------------------------------------------------------

typedef __attribute__((ext_vector_type(16))) _Float16 v16h;
typedef __attribute__((ext_vector_type(8)))  float    v8f;
typedef unsigned int u32x4 __attribute__((ext_vector_type(4)));
typedef int          i32x4 __attribute__((ext_vector_type(4)));
typedef int          i32x8 __attribute__((ext_vector_type(8)));

#ifndef __has_builtin
#define __has_builtin(x) 0
#endif
#if defined(__HIP_DEVICE_COMPILE__) && \
    __has_builtin(__builtin_amdgcn_tensor_load_to_lds) && \
    __has_builtin(__builtin_amdgcn_s_wait_tensorcnt)
#define HAVE_TDM 1
#else
#define HAVE_TDM 0
#endif

#define H0 96
#define W0 160
#define HW0 (H0*W0)       // 15360
#define HW1 (48*80)       // 3840
#define HW2 (24*40)       // 960
#define HW2X (192*320)    // 61440
#define HW4X (384*640)    // 245760

__device__ __forceinline__ float apply_act(float x, int act) {
  switch (act) {
    case 1: return x > 0.f ? x : 0.f;                 // relu
    case 2: return 1.f / (1.f + __expf(-x));          // sigmoid
    case 3: return tanhf(x);                          // tanh
    case 4: return x > 0.f ? x : 0.01f * x;           // leaky relu
    default: return x;
  }
}

#if HAVE_TDM
// Issue a TDM 2D tile load: tile_d0 contiguous elements (f32) per row,
// tile_d1 rows with row stride stride0 (elements). Zero-fills past
// (td0, td1) tensor extent per ISA 8.2. One instruction, tracked by TENSORcnt.
__device__ __forceinline__ void tdm_load_2d(const float* g, unsigned lds_off,
                                            unsigned tile_d0, unsigned tile_d1,
                                            unsigned td0, unsigned td1,
                                            unsigned stride0)
{
  unsigned long long ga = (unsigned long long)(size_t)g;
  u32x4 g0;
  g0[0] = 1u;                                             // count=1, user desc
  g0[1] = lds_off;                                        // LDS byte address
  g0[2] = (unsigned)ga;                                   // global_addr[31:0]
  g0[3] = (unsigned)((ga >> 32) & 0x1ffffffull)           // global_addr[56:32]
        | (2u << 30);                                     // type=2 ("image")
  i32x8 g1;
  g1[0] = (int)(2u << 16);                                // data_size=4B
  g1[1] = (int)((td0 & 0xffffu) << 16);                   // tensor_dim0[15:0]
  g1[2] = (int)((td0 >> 16) | ((td1 & 0xffffu) << 16));   // td0[31:16]|td1[15:0]
  g1[3] = (int)((td1 >> 16) | ((tile_d0 & 0xffffu) << 16));// td1[31:16]|tile0
  g1[4] = (int)(tile_d1 & 0xffffu);                       // tile_dim1 (tile2=0)
  g1[5] = (int)stride0;                                   // dim0 stride[31:0]
  g1[6] = 0;                                              // stride[47:32], d1s
  g1[7] = 0;
  i32x4 gz = {0, 0, 0, 0};
#if defined(__clang_major__) && (__clang_major__ >= 23)
  i32x8 gz8 = {0, 0, 0, 0, 0, 0, 0, 0};
  __builtin_amdgcn_tensor_load_to_lds(g0, g1, gz, gz, gz8, 0);
#else
  __builtin_amdgcn_tensor_load_to_lds(g0, g1, gz, gz, 0);
#endif
}
#endif

// Build one 16x32 f16 A fragment (ISA layout: row = lane%16, element e ->
// K = ((e<8)?e:e+8) + 8*half). The 16 K values form two contiguous runs of 8,
// so we decompose K->(ci,ky,kx) once per run and carry-propagate.
__device__ __forceinline__ v16h build_a_frag(
    const float* __restrict__ in0, int c0,
    const float* __restrict__ in1, int c1,
    const float* __restrict__ in2, int c2,
    int kbase, int half, int Ktot, int kk2, int kdim, int pad,
    int h, int w, int HW, bool rowok, int py, int px)
{
  v16h a;
#pragma unroll
  for (int run = 0; run < 2; ++run) {
    const int kst = kbase + (run ? 16 : 0) + 8 * half;
    int ci, ky, kx;
    if (kk2 == 1) { ci = kst; ky = 0; kx = 0; }
    else {
      ci = kst / kk2;
      int rem = kst - ci * kk2;
      ky = rem / kdim;
      kx = rem - ky * kdim;
    }
#pragma unroll
    for (int e = 0; e < 8; ++e) {
      float v = 0.0f;
      if ((kst + e) < Ktot && rowok) {
        int iy = py - pad + ky;
        int ix = px - pad + kx;
        if (iy >= 0 && iy < h && ix >= 0 && ix < w) {
          const float* src;
          int cl = ci;
          if (cl < c0)            { src = in0; }
          else if (cl < c0 + c1)  { src = in1; cl -= c0; }
          else                    { src = in2; cl -= (c0 + c1); }
          v = src[(size_t)cl * HW + (size_t)iy * w + ix];
        }
      }
      a[run * 8 + e] = (_Float16)v;
      ++kx;                       // carry-propagate K -> (ci,ky,kx)
      if (kx == kdim) { kx = 0; ++ky; if (ky == kdim) { ky = 0; ++ci; } }
    }
  }
  return a;
}

// ---------------------------------------------------------------------------
// Generic stride-1 "same" conv as implicit GEMM on WMMA.
//   M = output pixel, N = output channel, K = cin*kh*kw.
//   Input may be a virtual concat of up to 3 tensors (c0,c1,c2 channels).
// Block = 128 threads = 4 waves. Each wave owns a 32-pixel M-tile (2 A frags);
// the block shares a 64-channel N-block whose 32-K weight chunk is TDM-loaded
// into LDS (double buffered), LDS layout [n (64)][k (32)] f32.
// ---------------------------------------------------------------------------
__global__ __launch_bounds__(128)
void conv_wmma_kernel(const float* __restrict__ in0, int c0,
                      const float* __restrict__ in1, int c1,
                      const float* __restrict__ in2, int c2,
                      const float* __restrict__ wt,   // (cout, cin, k, k)
                      const float* __restrict__ bias, // (cout)
                      float* __restrict__ out,        // (cout, h, w)
                      int cout, int h, int w, int kdim, int pad, int act)
{
  __shared__ float lds_w[2][64 * 32];

  const int HW   = h * w;
  const int cin  = c0 + c1 + c2;
  const int kk2  = kdim * kdim;
  const int Ktot = cin * kk2;

  const int lane = threadIdx.x & 31;
  const int wave = threadIdx.x >> 5;
  const int half = lane >> 4;
  const int r    = lane & 15;

  const int m0 = (blockIdx.x * 4 + wave) * 32;
  const int n0 = blockIdx.y * 64;

  const bool ok0 = (m0 + r) < HW;          // subtile 0 rows
  const bool ok1 = (m0 + 16 + r) < HW;     // subtile 1 rows
  const int row0 = ok0 ? (m0 + r) : (HW - 1);
  const int row1 = ok1 ? (m0 + 16 + r) : (HW - 1);
  const int py0 = row0 / w, px0 = row0 - py0 * w;
  const int py1 = row1 / w, px1 = row1 - py1 * w;

  const int nchunks = (Ktot + 31) / 32;
  v8f acc[2][4] = {};

#if HAVE_TDM
  if (wave == 0) {
    tdm_load_2d(wt + (size_t)n0 * Ktot,
                (unsigned)(size_t)&lds_w[0][0],
                /*tile*/32, 64, /*tensor*/(unsigned)Ktot, (unsigned)(cout - n0),
                (unsigned)Ktot);
  }
#endif

  for (int ch = 0; ch < nchunks; ++ch) {
    const int kbase = ch * 32;
    __syncthreads();   // previous chunk's consumers done before buffer reuse
#if HAVE_TDM
    if (wave == 0) {
      if (ch + 1 < nchunks) {
        tdm_load_2d(wt + (size_t)n0 * Ktot + (kbase + 32),
                    (unsigned)(size_t)&lds_w[(ch + 1) & 1][0],
                    32, 64, (unsigned)(Ktot - (kbase + 32)),
                    (unsigned)(cout - n0), (unsigned)Ktot);
        __builtin_amdgcn_s_wait_tensorcnt(1);  // chunk ch done, ch+1 in flight
      } else {
        __builtin_amdgcn_s_wait_tensorcnt(0);
      }
    }
#else
    {
      float* dstb = &lds_w[ch & 1][0];
      for (int t = threadIdx.x; t < 64 * 32; t += 128) {
        int nn = t >> 5, kk = t & 31;
        int k = kbase + kk, co = n0 + nn;
        dstb[t] = (k < Ktot && co < cout) ? wt[(size_t)co * Ktot + k] : 0.f;
      }
      if (ch + 1 < nchunks)
        __builtin_prefetch(wt + (size_t)n0 * Ktot + (kbase + 32), 0, 1);
    }
#endif
    __syncthreads();

    v16h a0 = build_a_frag(in0, c0, in1, c1, in2, c2, kbase, half, Ktot,
                           kk2, kdim, pad, h, w, HW, ok0, py0, px0);
    v16h a1 = build_a_frag(in0, c0, in1, c1, in2, c2, kbase, half, Ktot,
                           kk2, kdim, pad, h, w, HW, ok1, py1, px1);

    const float* bb = &lds_w[ch & 1][0];
#pragma unroll
    for (int s = 0; s < 4; ++s) {
      v16h b;   // B layout: element e -> K = ((e<8)?e:e+8) + 8*half; N=lane%16
#pragma unroll
      for (int e = 0; e < 16; ++e)
        b[e] = (_Float16)bb[(s * 16 + r) * 32 + (((e < 8) ? e : e + 8) + 8 * half)];
      acc[0][s] = __builtin_amdgcn_wmma_f32_16x16x32_f16(
          false, a0, false, b, (short)0, acc[0][s], false, false);
      acc[1][s] = __builtin_amdgcn_wmma_f32_16x16x32_f16(
          false, a1, false, b, (short)0, acc[1][s], false, false);
    }
  }

  // D layout: vgpr j -> M = j + 8*half, N = lane%16.
#pragma unroll
  for (int u = 0; u < 2; ++u) {
#pragma unroll
    for (int s = 0; s < 4; ++s) {
      int co = n0 + s * 16 + r;
      float bv = (co < cout) ? bias[co] : 0.0f;
#pragma unroll
      for (int j = 0; j < 8; ++j) {
        int m = m0 + u * 16 + j + 8 * half;
        if (co < cout && m < HW) {
          out[(size_t)co * HW + m] = apply_act(acc[u][s][j] + bv, act);
        }
      }
    }
  }
}

// ---------------------------------------------------------------------------
// Correlation: corr[(h*W+wl), wr] = sum_c left[c,h,wl]*right[c,h,wr] / sqrt(128)
// Per-row batched 160x160x128 GEMM. B tile (32 k x 64 wr) is a natural TDM 2D
// tile: contiguous along wr, row stride HW0 between channels.
// ---------------------------------------------------------------------------
__global__ __launch_bounds__(128)
void corr_wmma_kernel(const float* __restrict__ left,
                      const float* __restrict__ right,
                      float* __restrict__ corr)
{
  __shared__ float lds_r[2][32 * 64];
  const int lane = threadIdx.x & 31;
  const int wave = threadIdx.x >> 5;
  const int half = lane >> 4;
  const int r    = lane & 15;

  const int hh = blockIdx.z;
  const int n0 = blockIdx.y * 64;
  const int m0 = (blockIdx.x * 4 + wave) * 16;
  const int mm = (m0 + r < W0) ? (m0 + r) : (W0 - 1);

  v8f acc[4] = {};

#if HAVE_TDM
  if (wave == 0) {
    tdm_load_2d(right + (size_t)hh * W0 + n0,
                (unsigned)(size_t)&lds_r[0][0],
                /*tile*/64, 32, /*tensor*/(unsigned)(W0 - n0), 128u,
                (unsigned)HW0);
  }
#endif

  for (int ch = 0; ch < 4; ++ch) {        // K = 128
    const int kbase = ch * 32;
    __syncthreads();
#if HAVE_TDM
    if (wave == 0) {
      if (ch + 1 < 4) {
        tdm_load_2d(right + (size_t)(kbase + 32) * HW0 + (size_t)hh * W0 + n0,
                    (unsigned)(size_t)&lds_r[(ch + 1) & 1][0],
                    64, 32, (unsigned)(W0 - n0), (unsigned)(128 - (kbase + 32)),
                    (unsigned)HW0);
        __builtin_amdgcn_s_wait_tensorcnt(1);
      } else {
        __builtin_amdgcn_s_wait_tensorcnt(0);
      }
    }
#else
    {
      float* dstb = &lds_r[ch & 1][0];
      for (int t = threadIdx.x; t < 32 * 64; t += 128) {
        int kk = t >> 6, nn = t & 63;
        int n = n0 + nn;
        dstb[t] = (n < W0)
            ? right[(size_t)(kbase + kk) * HW0 + (size_t)hh * W0 + n] : 0.f;
      }
    }
#endif
    __syncthreads();

    v16h a;
#pragma unroll
    for (int e = 0; e < 16; ++e) {
      int klocal = ((e < 8) ? e : e + 8) + 8 * half;
      a[e] = (_Float16)left[(size_t)(kbase + klocal) * HW0 + (size_t)hh * W0 + mm];
    }
    const float* bb = &lds_r[ch & 1][0];
#pragma unroll
    for (int s = 0; s < 4; ++s) {
      v16h b;   // LDS layout [k][n=64]
#pragma unroll
      for (int e = 0; e < 16; ++e)
        b[e] = (_Float16)bb[(((e < 8) ? e : e + 8) + 8 * half) * 64 + (s * 16 + r)];
      acc[s] = __builtin_amdgcn_wmma_f32_16x16x32_f16(
          false, a, false, b, (short)0, acc[s], false, false);
    }
  }

  const float scale = 0.08838834764831845f;  // 1/sqrt(128)
#pragma unroll
  for (int s = 0; s < 4; ++s) {
#pragma unroll
    for (int j = 0; j < 8; ++j) {
      int m = m0 + j + 8 * half;
      int n = n0 + s * 16 + r;
      if (m < W0 && n < W0)
        corr[((size_t)hh * W0 + m) * W0 + n] = acc[s][j] * scale;
    }
  }
}

// ---------------------------------------------------------------------------
// geo_lookup: build 522-channel feature directly from the NCDHW cost volume
// and the correlation matrix (no materialized pyramid transpose).
// thread = (tap t in 0..8, pixel n); channel order g0(252) c0(9) g1(252) c1(9)
// ---------------------------------------------------------------------------
__global__ void geo_lookup_kernel(const float* __restrict__ vol,   // (28,104,96,160)
                                  const float* __restrict__ corr,  // (HW0, 160)
                                  const float* __restrict__ disp,  // (HW0)
                                  float* __restrict__ out)         // (522, HW0)
{
  int idx = blockIdx.x * blockDim.x + threadIdx.x;
  if (idx >= HW0 * 9) return;
  const int t = idx / HW0;
  const int n = idx - t * HW0;
  const float d   = disp[n];
  const int   ww  = n % W0;
  const float dxt = (float)(t - 4);
  const float* cb = corr + (size_t)n * W0;

  // level-0 geo (D = 104)
  {
    float xg = d + dxt;
    float x0 = floorf(xg);
    float f  = xg - x0;
    int   i0 = (int)x0;
    bool ok0 = (i0 >= 0 && i0 < 104);
    bool ok1 = (i0 + 1 >= 0 && i0 + 1 < 104);
    for (int c = 0; c < 28; ++c) {
      const float* base = vol + (size_t)c * 104 * HW0 + n;
      float a = ok0 ? base[(size_t)i0 * HW0] : 0.f;
      float b = ok1 ? base[(size_t)(i0 + 1) * HW0] : 0.f;
      out[(size_t)(c * 9 + t) * HW0 + n] = a * (1.f - f) + b * f;
    }
  }
  // level-0 corr (D = 160)
  {
    float xc = ((float)ww - d) + dxt;
    float x0 = floorf(xc);
    float f  = xc - x0;
    int   i0 = (int)x0;
    float a = (i0 >= 0 && i0 < W0) ? cb[i0] : 0.f;
    float b = (i0 + 1 >= 0 && i0 + 1 < W0) ? cb[i0 + 1] : 0.f;
    out[(size_t)(252 + t) * HW0 + n] = a * (1.f - f) + b * f;
  }
  // level-1 geo (D = 52, avg-pooled pairs on the fly)
  {
    float xg = d * 0.5f + dxt;
    float x0 = floorf(xg);
    float f  = xg - x0;
    int   i0 = (int)x0;
    for (int c = 0; c < 28; ++c) {
      const float* base = vol + (size_t)c * 104 * HW0 + n;
      float a = 0.f, b = 0.f;
      if (i0 >= 0 && i0 < 52)
        a = 0.5f * (base[(size_t)(2 * i0) * HW0] + base[(size_t)(2 * i0 + 1) * HW0]);
      if (i0 + 1 >= 0 && i0 + 1 < 52)
        b = 0.5f * (base[(size_t)(2 * i0 + 2) * HW0] + base[(size_t)(2 * i0 + 3) * HW0]);
      out[(size_t)(261 + c * 9 + t) * HW0 + n] = a * (1.f - f) + b * f;
    }
  }
  // level-1 corr (D = 80)
  {
    float xc = ((float)ww - d) * 0.5f + dxt;
    float x0 = floorf(xc);
    float f  = xc - x0;
    int   i0 = (int)x0;
    float a = (i0 >= 0 && i0 < 80) ? 0.5f * (cb[2 * i0] + cb[2 * i0 + 1]) : 0.f;
    float b = (i0 + 1 >= 0 && i0 + 1 < 80) ? 0.5f * (cb[2 * i0 + 2] + cb[2 * i0 + 3]) : 0.f;
    out[(size_t)(513 + t) * HW0 + n] = a * (1.f - f) + b * f;
  }
}

// avg_pool2d(k=3, s=2, p=1), count_include_pad
__global__ void pool2x_kernel(const float* __restrict__ in, float* __restrict__ out,
                              int C, int ih, int iw)
{
  int oh = ih / 2, ow = iw / 2;
  int idx = blockIdx.x * blockDim.x + threadIdx.x;
  if (idx >= C * oh * ow) return;
  int x = idx % ow;
  int y = (idx / ow) % oh;
  int c = idx / (ow * oh);
  float s = 0.f;
  for (int dy = -1; dy <= 1; ++dy)
    for (int dx = -1; dx <= 1; ++dx) {
      int iy = 2 * y + dy, ix = 2 * x + dx;
      if (iy >= 0 && iy < ih && ix >= 0 && ix < iw)
        s += in[((size_t)c * ih + iy) * iw + ix];
    }
  out[idx] = s * (1.f / 9.f);
}

// bilinear, align_corners=True
__global__ void interp_ac_kernel(const float* __restrict__ in, float* __restrict__ out,
                                 int C, int ih, int iw, int oh, int ow)
{
  int idx = blockIdx.x * blockDim.x + threadIdx.x;
  if (idx >= C * oh * ow) return;
  int x = idx % ow;
  int y = (idx / ow) % oh;
  int c = idx / (ow * oh);
  float ys = (oh > 1) ? (float)y * (float)(ih - 1) / (float)(oh - 1) : 0.f;
  float xs = (ow > 1) ? (float)x * (float)(iw - 1) / (float)(ow - 1) : 0.f;
  int y0 = (int)floorf(ys); if (y0 < 0) y0 = 0; if (y0 > ih - 1) y0 = ih - 1;
  int x0 = (int)floorf(xs); if (x0 < 0) x0 = 0; if (x0 > iw - 1) x0 = iw - 1;
  int y1 = (y0 + 1 < ih) ? y0 + 1 : ih - 1;
  int x1 = (x0 + 1 < iw) ? x0 + 1 : iw - 1;
  float wy = ys - (float)y0;
  float wx = xs - (float)x0;
  const float* p = in + (size_t)c * ih * iw;
  float tl = p[(size_t)y0 * iw + x0], tr = p[(size_t)y0 * iw + x1];
  float bl = p[(size_t)y1 * iw + x0], br = p[(size_t)y1 * iw + x1];
  float top = tl * (1.f - wx) + tr * wx;
  float bot = bl * (1.f - wx) + br * wx;
  out[idx] = top * (1.f - wy) + bot * wy;
}

// ConvTranspose2d(k=4, s=2, p=1) as direct gather (kernel pre-flipped, OIHW)
__global__ void deconv_k4s2_kernel(const float* __restrict__ in,
                                   const float* __restrict__ wt,
                                   const float* __restrict__ bias,
                                   float* __restrict__ out,
                                   int cin, int cout, int ih, int iw, int act)
{
  int oh = 2 * ih, ow = 2 * iw;
  int idx = blockIdx.x * blockDim.x + threadIdx.x;
  if (idx >= cout * oh * ow) return;
  int x = idx % ow;
  int y = (idx / ow) % oh;
  int o = idx / (ow * oh);
  float s = bias[o];
  for (int ky = 0; ky < 4; ++ky) {
    int dy = y - 2 + ky;
    if (dy < 0 || (dy & 1) || (dy >> 1) >= ih) continue;
    int sy = dy >> 1;
    for (int kx = 0; kx < 4; ++kx) {
      int dxx = x - 2 + kx;
      if (dxx < 0 || (dxx & 1) || (dxx >> 1) >= iw) continue;
      int sx = dxx >> 1;
      const float* wrow = wt + (size_t)o * cin * 16 + ky * 4 + kx;
      const float* irow = in + (size_t)sy * iw + sx;
      for (int i = 0; i < cin; ++i)
        s += irow[(size_t)i * ih * iw] * wrow[(size_t)i * 16];
    }
  }
  out[idx] = apply_act(s, act);
}

__global__ void softmax9_kernel(float* __restrict__ w, int hw)
{
  int n = blockIdx.x * blockDim.x + threadIdx.x;
  if (n >= hw) return;
  float m = -1e30f;
  for (int c = 0; c < 9; ++c) m = fmaxf(m, w[(size_t)c * hw + n]);
  float e[9], s = 0.f;
  for (int c = 0; c < 9; ++c) { e[c] = __expf(w[(size_t)c * hw + n] - m); s += e[c]; }
  float inv = 1.f / s;
  for (int c = 0; c < 9; ++c) w[(size_t)c * hw + n] = e[c] * inv;
}

// context_upsample: 3x3 unfold of disp*4 (pad 1), nearest x4, convex blend
__global__ void ctx_upsample_kernel(const float* __restrict__ disp,  // (96,160)
                                    const float* __restrict__ wts,   // (9,384,640)
                                    float* __restrict__ out)         // (384,640)
{
  int idx = blockIdx.x * blockDim.x + threadIdx.x;
  if (idx >= HW4X) return;
  int x = idx % 640;
  int y = idx / 640;
  int ly = y >> 2, lx = x >> 2;
  float acc = 0.f;
  for (int ti = 0; ti < 3; ++ti)
    for (int tj = 0; tj < 3; ++tj) {
      int sy = ly + ti - 1, sx = lx + tj - 1;
      float dv = (sy >= 0 && sy < H0 && sx >= 0 && sx < W0) ? disp[sy * W0 + sx] * 4.f : 0.f;
      acc += dv * wts[(size_t)(ti * 3 + tj) * HW4X + idx];
    }
  out[idx] = acc;
}

// elementwise helpers
__global__ void ew_mul_kernel(float* a, const float* __restrict__ b, int n)
{ int i = blockIdx.x * blockDim.x + threadIdx.x; if (i < n) a[i] *= b[i]; }

__global__ void ew_gruout_kernel(float* __restrict__ dst, const float* __restrict__ z,
                                 const float* __restrict__ h, const float* __restrict__ q, int n)
{ int i = blockIdx.x * blockDim.x + threadIdx.x;
  if (i < n) dst[i] = (1.f - z[i]) * h[i] + z[i] * q[i]; }

__global__ void ew_mix_kernel(float* __restrict__ out, const float* __restrict__ att,
                              const float* __restrict__ hs, const float* __restrict__ hl,
                              int hw, int C)
{ int i = blockIdx.x * blockDim.x + threadIdx.x;
  if (i < C * hw) { float a = att[i % hw]; out[i] = a * hs[i] + (1.f - a) * hl[i]; } }

__global__ void ew_add_kernel(float* a, const float* __restrict__ b, int n)
{ int i = blockIdx.x * blockDim.x + threadIdx.x; if (i < n) a[i] += b[i]; }

// ---------------------------------------------------------------------------
// Host orchestration
// ---------------------------------------------------------------------------
extern "C" void kernel_launch(void* const* d_in, const int* in_sizes, int n_in,
                              void* d_out, int out_size, void* d_ws, size_t ws_size,
                              hipStream_t stream) {
  (void)in_sizes; (void)n_in; (void)out_size; (void)ws_size;

  const float* init_disp = (const float*)d_in[0];
  const float* left      = (const float*)d_in[1];
  const float* right     = (const float*)d_in[2];
  const float* volume    = (const float*)d_in[3];
  const float* inp0      = (const float*)d_in[4];
  const float* inp1      = (const float*)d_in[5];
  const float* inp2      = (const float*)d_in[6];
  const float* net0_in   = (const float*)d_in[7];
  const float* net1_in   = (const float*)d_in[8];
  const float* net2_in   = (const float*)d_in[9];
  const float* att0      = (const float*)d_in[10];
  const float* att1      = (const float*)d_in[11];
  const float* att2      = (const float*)d_in[12];
  const float* stem2x    = (const float*)d_in[13];

  // Param arrays in setup_inputs() insertion order; each conv = (w, b).
  enum {
    P = 14,
    ENC_C1 = P + 0, ENC_C2 = P + 2, ENC_D1 = P + 4, ENC_D2 = P + 6, ENC_CONV = P + 8,
    G16 = P + 10, G08 = P + 26, G04 = P + 42,      // gru block: +0 conv0, +2 conv1,
    DH1 = P + 58, DH2 = P + 60, MASK = P + 62,     //   +4 s.z, +6 s.r, +8 s.q,
    SPX2_UP = P + 64, SPX2_CONV = P + 66, SPX_GRU = P + 68  //   +10 l.z, +12 l.r, +14 l.q
  };

  // -------- workspace (bump allocation, floats) --------
  float* base = (float*)d_ws;
  size_t off = 0;
  auto alloc = [&](size_t n) { float* p = base + off; off += n; return p; };

  float* corr  = alloc((size_t)HW0 * W0);
  float* disp  = alloc(HW0);
  float* delta = alloc(HW0);
  float* netA0 = alloc((size_t)128 * HW0);
  float* netB0 = alloc((size_t)128 * HW0);
  float* netA1 = alloc((size_t)128 * HW1);
  float* netB1 = alloc((size_t)128 * HW1);
  float* netA2 = alloc((size_t)128 * HW2);
  float* netB2 = alloc((size_t)128 * HW2);
  float* pool0 = alloc((size_t)128 * HW1);
  float* pool1 = alloc((size_t)128 * HW2);
  float* upn2  = alloc((size_t)128 * HW1);
  float* upn1  = alloc((size_t)128 * HW0);
  float* geo   = alloc((size_t)522 * HW0);
  float* motion= alloc((size_t)128 * HW0);
  float* e0 = alloc((size_t)64 * HW0);
  float* e1 = alloc((size_t)64 * HW0);
  float* e2 = alloc((size_t)64 * HW0);
  float* e3 = alloc((size_t)64 * HW0);
  float* X0 = alloc((size_t)384 * HW0);  // gru x / dh1 scratch
  float* X1 = alloc((size_t)384 * HW0);
  float* Zb = alloc((size_t)128 * HW0);
  float* Rb = alloc((size_t)128 * HW0);
  float* Qb = alloc((size_t)128 * HW0);
  float* HS = alloc((size_t)128 * HW0);
  float* HL = alloc((size_t)128 * HW0);
  float* maskf = alloc((size_t)32 * HW0);
  // spx head runs after the iteration loop: alias onto free scratch
  float* spxX = X0;          // 32*HW2X  <= 384*HW0
  float* spxC = X1;          // 64*HW2X  <= 384*HW0
  float* spxW = geo;         // 9*HW4X   <= 522*HW0

  auto blocks = [](int n) { return (unsigned)((n + 255) / 256); };

  auto conv = [&](const float* a, int ca, const float* b2, int cb,
                  const float* c3, int cc, int widx, float* o,
                  int cout, int h, int w, int k, int pad, int act) {
    dim3 grid((unsigned)((h * w + 127) / 128), (unsigned)((cout + 63) / 64));
    conv_wmma_kernel<<<grid, 128, 0, stream>>>(
        a, ca, b2, cb, c3, cc,
        (const float*)d_in[widx], (const float*)d_in[widx + 1],
        o, cout, h, w, k, pad, act);
  };

  auto sel_gru = [&](int g, const float* attp, const float* h,
                     const float* x0, int cx0, const float* x1p, int cx1,
                     const float* x2, int cx2, float* hout, int ih, int iw) {
    const int ci = cx0 + cx1 + cx2;
    const int hw = ih * iw;
    conv(x0, cx0, x1p, cx1, x2, cx2, g + 0, X0, ci, ih, iw, 1, 0, 1);         // conv0+relu
    conv(X0, ci, (const float*)nullptr, 0, (const float*)nullptr, 0,
         g + 2, X1, ci, ih, iw, 3, 1, 1);                                      // conv1+relu
    for (int bi = 0; bi < 2; ++bi) {                                           // 0=small,1=large
      int k = bi ? 3 : 1, pad = bi ? 1 : 0;
      int zo = g + (bi ? 10 : 4);
      float* dst = bi ? HL : HS;
      conv(h, 128, X1, ci, (const float*)nullptr, 0, zo + 0, Zb, 128, ih, iw, k, pad, 2);
      conv(h, 128, X1, ci, (const float*)nullptr, 0, zo + 2, Rb, 128, ih, iw, k, pad, 2);
      ew_mul_kernel<<<blocks(128 * hw), 256, 0, stream>>>(Rb, h, 128 * hw);    // r*h
      conv(Rb, 128, X1, ci, (const float*)nullptr, 0, zo + 4, Qb, 128, ih, iw, k, pad, 3);
      ew_gruout_kernel<<<blocks(128 * hw), 256, 0, stream>>>(dst, Zb, h, Qb, 128 * hw);
    }
    ew_mix_kernel<<<blocks(128 * hw), 256, 0, stream>>>(hout, attp, HS, HL, hw, 128);
  };

  // -------- init state --------
  hipMemcpyAsync(disp,  init_disp, (size_t)HW0 * 4,        hipMemcpyDeviceToDevice, stream);
  hipMemcpyAsync(netA0, net0_in,   (size_t)128 * HW0 * 4,  hipMemcpyDeviceToDevice, stream);
  hipMemcpyAsync(netA1, net1_in,   (size_t)128 * HW1 * 4,  hipMemcpyDeviceToDevice, stream);
  hipMemcpyAsync(netA2, net2_in,   (size_t)128 * HW2 * 4,  hipMemcpyDeviceToDevice, stream);

  // -------- correlation (once) --------
  corr_wmma_kernel<<<dim3(3, 3, 96), 128, 0, stream>>>(left, right, corr);

  float* cur0 = netA0; float* nxt0 = netB0;
  float* cur1 = netA1; float* nxt1 = netB1;
  float* cur2 = netA2; float* nxt2 = netB2;

  for (int it = 0; it < 2; ++it) {
    // geo/corr feature lookup
    geo_lookup_kernel<<<blocks(HW0 * 9), 256, 0, stream>>>(volume, corr, disp, geo);

    // motion encoder
    conv(geo, 522, (const float*)nullptr, 0, (const float*)nullptr, 0, ENC_C1, e0, 64, H0, W0, 1, 0, 1);
    conv(e0, 64,  (const float*)nullptr, 0, (const float*)nullptr, 0, ENC_C2, e1, 64, H0, W0, 3, 1, 1);
    conv(disp, 1, (const float*)nullptr, 0, (const float*)nullptr, 0, ENC_D1, e2, 64, H0, W0, 7, 3, 1);
    conv(e2, 64,  (const float*)nullptr, 0, (const float*)nullptr, 0, ENC_D2, e3, 64, H0, W0, 3, 1, 1);
    conv(e1, 64, e3, 64, (const float*)nullptr, 0, ENC_CONV, motion, 127, H0, W0, 3, 1, 1);
    hipMemcpyAsync(motion + (size_t)127 * HW0, disp, (size_t)HW0 * 4,
                   hipMemcpyDeviceToDevice, stream);                // motion ch 127 = disp

    // pooled hidden states (old nets)
    pool2x_kernel<<<blocks(128 * HW1), 256, 0, stream>>>(cur0, pool0, 128, H0, W0);
    pool2x_kernel<<<blocks(128 * HW2), 256, 0, stream>>>(cur1, pool1, 128, 48, 80);

    // gru16 -> gru08 -> gru04
    sel_gru(G16, att2, cur2, inp2, 128, pool1, 128, (const float*)nullptr, 0, nxt2, 24, 40);
    interp_ac_kernel<<<blocks(128 * HW1), 256, 0, stream>>>(nxt2, upn2, 128, 24, 40, 48, 80);
    sel_gru(G08, att1, cur1, inp1, 128, pool0, 128, upn2, 128, nxt1, 48, 80);
    interp_ac_kernel<<<blocks(128 * HW0), 256, 0, stream>>>(nxt1, upn1, 128, 48, 80, H0, W0);
    sel_gru(G04, att0, cur0, inp0, 128, motion, 128, upn1, 128, nxt0, H0, W0);

    // disp head + mask feature
    conv(nxt0, 128, (const float*)nullptr, 0, (const float*)nullptr, 0, DH1, X0, 256, H0, W0, 3, 1, 1);
    conv(X0, 256,   (const float*)nullptr, 0, (const float*)nullptr, 0, DH2, delta, 1, H0, W0, 3, 1, 0);
    conv(nxt0, 128, (const float*)nullptr, 0, (const float*)nullptr, 0, MASK, maskf, 32, H0, W0, 3, 1, 1);
    ew_add_kernel<<<blocks(HW0), 256, 0, stream>>>(disp, delta, HW0);

    float* t;
    t = cur0; cur0 = nxt0; nxt0 = t;
    t = cur1; cur1 = nxt1; nxt1 = t;
    t = cur2; cur2 = nxt2; nxt2 = t;
  }

  // -------- convex upsample head --------
  deconv_k4s2_kernel<<<blocks(32 * HW2X), 256, 0, stream>>>(
      maskf, (const float*)d_in[SPX2_UP], (const float*)d_in[SPX2_UP + 1],
      spxX, 32, 32, H0, W0, 4);                                    // lrelu
  conv(spxX, 32, stem2x, 32, (const float*)nullptr, 0, SPX2_CONV, spxC, 64, 192, 320, 3, 1, 4);
  deconv_k4s2_kernel<<<blocks(9 * HW4X), 256, 0, stream>>>(
      spxC, (const float*)d_in[SPX_GRU], (const float*)d_in[SPX_GRU + 1],
      spxW, 64, 9, 192, 320, 0);
  softmax9_kernel<<<blocks(HW4X), 256, 0, stream>>>(spxW, HW4X);
  ctx_upsample_kernel<<<blocks(HW4X), 256, 0, stream>>>(disp, spxW, (float*)d_out);
}
